// Model_15126874816812
// MI455X (gfx1250) — matrix operand order repaired and checked by reading, not compiled, not run on hardware
//
#include <hip/hip_runtime.h>
#include <hip/hip_bf16.h>

// ---------------------------------------------------------------------------
// VQ-VAE forward for gfx1250. All convs / GEMMs via v_wmma_f32_16x16x32_f16.
// Activations: NHWC f16 (channels padded to multiples of 32, compile-time).
// Per block (2 waves, 64 output px, 4 M-tiles = 2 per wave): for each tap the
// full 64px x CINP B-row is staged to LDS once (OOB taps read a zero page),
// then the channel loop fully unrolls: each K-step = 2 A loads + 4 B fragment
// pairs (ds_load_b128) feeding 8 v_wmma (B fragments reused across M-tiles).
// ---------------------------------------------------------------------------

typedef __attribute__((ext_vector_type(16))) _Float16 v16h;
typedef __attribute__((ext_vector_type(8)))  _Float16 v8h;
typedef __attribute__((ext_vector_type(8)))  float    v8f;

#define NT 4  // 16-wide N tiles per wave
#define MT 2  // 16-tall M tiles per wave (B fragments reused across them)

// Epilogue modes:
//   0: BN(eval)+ReLU -> outh (NHWC f16)
//   1: ReLU(acc+bias) -> outh
//   2: acc+bias       -> outh
//   3: sigmoid(acc+bias) -> outf (NCHW f32), guarded m<Cout
//   4: raw acc        -> outf[n*Cout+m]   (VQ Gram matrix)
template <int KH, int KW, int STRIDE, int PAD, int TR, int MODE, int CINP>
__global__ __launch_bounds__(64) void k_conv_wmma(
    const _Float16* __restrict__ act,   // NHWC f16 (Bn, inH, inW, CINP)
    const _Float16* __restrict__ wrow,  // (Mpad x CINP*KH*KW) f16, K=(ky*KW+kx)*CINP+cin
    const _Float16* __restrict__ zbuf,  // >= CINP zero halfs (OOB tap source)
    const float* __restrict__ bias,
    const float* __restrict__ bng, const float* __restrict__ bnb,
    const float* __restrict__ bnm, const float* __restrict__ bnv,
    _Float16* __restrict__ outh, float* __restrict__ outf,
    int inH, int inW,
    int outH, int outW, int Cout, int Mpad)
{
    __shared__ __align__(16) _Float16 sB[64 * CINP];  // 64 px x CINP ch

    constexpr int Ktot = CINP * KH * KW;
    const int lane = threadIdx.x;
    const int g  = lane >> 4;          // lane group
    const int ln = lane & 15;
    const int nBase0 = blockIdx.x * (16 * NT);

    // M tile-pair for this wave; tail pairs recompute last pair, stores masked
    const int mPairs = Mpad >> 5;      // Mpad is a multiple of 32
    int mp = blockIdx.y * 2 + threadIdx.y;
    const int mpOrig = mp;
    if (mp >= mPairs) mp = mPairs - 1;
    const int mBase = mp * 32;
    bool okU[MT];
#pragma unroll
    for (int u = 0; u < MT; ++u)
        okU[u] = (mpOrig * 32 + u * 16) < Mpad && (mpOrig < mPairs);

    // ---- staging role: thread -> pixel 0..63 (full CINP row each)
    const int tid = threadIdx.y * 32 + threadIdx.x;  // 0..63
    const int sp  = tid;
    int sb, soy, sox;
    {
        int n = nBase0 + sp;
        sox = n % outW; int tmp = n / outW;
        soy = tmp % outH; sb = tmp / outH;
    }

    // ---- compute role: per-lane output pixel coords per N tile
    int bA[NT], oyA[NT], oxA[NT];
#pragma unroll
    for (int t = 0; t < NT; ++t) {
        int n = nBase0 + t * 16 + ln;
        oxA[t] = n % outW; int tmp = n / outW;
        oyA[t] = tmp % outH; bA[t] = tmp / outH;
    }

    v8f acc[MT][NT];
#pragma unroll
    for (int u = 0; u < MT; ++u)
#pragma unroll
        for (int t = 0; t < NT; ++t)
#pragma unroll
            for (int r = 0; r < 8; ++r) acc[u][t][r] = 0.0f;

    const _Float16* wrU[MT];
#pragma unroll
    for (int u = 0; u < MT; ++u)
        wrU[u] = wrow + (size_t)(mBase + u * 16 + ln) * Ktot;

#pragma unroll
    for (int ky = 0; ky < KH; ++ky)
#pragma unroll
    for (int kx = 0; kx < KW; ++kx) {
        const int tap = ky * KW + kx;

        // staging source for this thread's pixel at this tap (zero page if OOB)
        const _Float16* src;
        {
            int iy, ix; bool valid;
            if (!TR) {
                iy = soy * STRIDE - PAD + ky;
                ix = sox * STRIDE - PAD + kx;
                valid = (iy >= 0) & (iy < inH) & (ix >= 0) & (ix < inW);
            } else {
                int ty = soy + PAD - ky;
                int tx = sox + PAD - kx;
                iy = ty / STRIDE; ix = tx / STRIDE;
                valid = (ty >= 0) & (tx >= 0) &
                        ((ty % STRIDE) == 0) & ((tx % STRIDE) == 0) &
                        (iy < inH) & (ix < inW);
            }
            src = valid ? act + (((size_t)sb * inH + iy) * inW + ix) * CINP : zbuf;
        }

        __syncthreads();   // previous tap's tile fully consumed
        // ---- stage full 64px x CINP row: one pixel row per thread
#pragma unroll
        for (int c = 0; c < CINP; c += 16) {
            const v8h d0 = *(const v8h*)(src + c);
            const v8h d1 = *(const v8h*)(src + c + 8);
            *(v8h*)(sB + sp * CINP + c)     = d0;
            *(v8h*)(sB + sp * CINP + c + 8) = d1;
        }
        __syncthreads();   // tile visible to both waves

#pragma unroll
        for (int c0 = 0; c0 < CINP; c0 += 32) {
            // ---- A fragments (one per M sub-tile): 16-bit A 16x32 layout
            v16h afrag[MT];
#pragma unroll
            for (int u = 0; u < MT; ++u) {
                const _Float16* wtap = wrU[u] + tap * CINP;
                const v8h lo = *(const v8h*)(wtap + c0 + g * 8);
                const v8h hi = *(const v8h*)(wtap + c0 + 16 + g * 8);
#pragma unroll
                for (int i = 0; i < 8; ++i) { afrag[u][i] = lo[i]; afrag[u][i + 8] = hi[i]; }
            }
#pragma unroll
            for (int t = 0; t < NT; ++t) {
                // B fragment: lane = col N; elem i<8 -> K = c-block + g*8 + i, i>=8 -> c-block + 16 + g*8 + (i-8)
                v16h bfrag;
                const v8h b0 = *(const v8h*)(sB + (t * 16 + ln) * CINP + c0 + g * 8);
                const v8h b1 = *(const v8h*)(sB + (t * 16 + ln) * CINP + c0 + 16 + g * 8);
#pragma unroll
                for (int i = 0; i < 8; ++i) { bfrag[i] = b0[i]; bfrag[i + 8] = b1[i]; }
#pragma unroll
                for (int u = 0; u < MT; ++u)
                    acc[u][t] = __builtin_amdgcn_wmma_f32_16x16x32_f16(
                        false, afrag[u], false, bfrag, (short)0, acc[u][t], false, false);
            }
        }
    }

    // epilogue: C/D layout: elem r at lane -> M = mBase + u*16 + r + g*8, N = ln
#pragma unroll
    for (int u = 0; u < MT; ++u) {
        if (!okU[u]) continue;
#pragma unroll
        for (int t = 0; t < NT; ++t) {
            const int n = nBase0 + t * 16 + ln;
            const int b = bA[t], oy = oyA[t], ox = oxA[t];
#pragma unroll
            for (int r = 0; r < 8; ++r) {
                const int m = mBase + u * 16 + r + g * 8;
                float v = acc[u][t][r];
                if (MODE == 0) {
                    float inv = bng[m] * rsqrtf(bnv[m] + 1e-5f);
                    float sh  = bnb[m] - bnm[m] * inv;
                    v = fmaxf((v + bias[m]) * inv + sh, 0.0f);
                    outh[(size_t)n * Cout + m] = (_Float16)v;
                } else if (MODE == 1) {
                    v = fmaxf(v + bias[m], 0.0f);
                    outh[(size_t)n * Cout + m] = (_Float16)v;
                } else if (MODE == 2) {
                    v = v + bias[m];
                    outh[(size_t)n * Cout + m] = (_Float16)v;
                } else if (MODE == 3) {
                    if (m < Cout) {
                        v = 1.0f / (1.0f + __expf(-(v + bias[m])));
                        outf[(((size_t)b * Cout + m) * outH + oy) * outW + ox] = v;
                    }
                } else {
                    outf[(size_t)n * Cout + m] = v;
                }
            }
        }
    }
}

// ---- pack conv weight (Cout,Cin,KH,KW) f32 -> (Mpad x Cinp*KH*KW) f16 rows
__global__ void k_pack_conv(const float* __restrict__ w, _Float16* __restrict__ dst,
                            int Cout, int Mpad, int Cin, int Cinp, int KH, int KW)
{
    int Ktot = Cinp * KH * KW;
    int id = blockIdx.x * blockDim.x + threadIdx.x;
    if (id >= Mpad * Ktot) return;
    int co = id / Ktot, k = id - co * Ktot;
    int tap = k / Cinp, cin = k - tap * Cinp;
    int ky = tap / KW, kx = tap - ky * KW;
    float v = 0.0f;
    if (co < Cout && cin < Cin) v = w[(((size_t)co * Cin + cin) * KH + ky) * KW + kx];
    dst[id] = (_Float16)v;
}

// ---- pack convT weight (Cin,Cout,KH,KW) f32 -> (Mpad x Cinp*KH*KW) f16 rows
__global__ void k_pack_trans(const float* __restrict__ w, _Float16* __restrict__ dst,
                             int Cout, int Mpad, int Cin, int Cinp, int KH, int KW)
{
    int Ktot = Cinp * KH * KW;
    int id = blockIdx.x * blockDim.x + threadIdx.x;
    if (id >= Mpad * Ktot) return;
    int co = id / Ktot, k = id - co * Ktot;
    int tap = k / Cinp, cin = k - tap * Cinp;
    int ky = tap / KW, kx = tap - ky * KW;
    float v = 0.0f;
    if (co < Cout && cin < Cin) v = w[(((size_t)cin * Cout + co) * KH + ky) * KW + kx];
    dst[id] = (_Float16)v;
}

// ---- x (NCHW f32, C=Cin) -> NHWC f16 with channel padding to Cinp (zeros)
__global__ void k_x_nhwc(const float* __restrict__ x, _Float16* __restrict__ dst,
                         int Bn, int Cin, int Cinp, int H, int W)
{
    size_t total = (size_t)Bn * H * W * Cinp;
    size_t id = blockIdx.x * (size_t)blockDim.x + threadIdx.x;
    if (id >= total) return;
    int c  = (int)(id % Cinp); size_t t = id / Cinp;
    int xw = (int)(t % W); t /= W;
    int yy = (int)(t % H); int b = (int)(t / H);
    float v = 0.0f;
    if (c < Cin) v = x[(((size_t)b * Cin + c) * H + yy) * W + xw];
    dst[id] = (_Float16)v;
}

// ---- zero-fill an f16 buffer (the OOB-tap zero page)
__global__ void k_zero_h(_Float16* __restrict__ dst, int n)
{
    int id = blockIdx.x * blockDim.x + threadIdx.x;
    if (id < n) dst[id] = (_Float16)0.0f;
}

// ---- codebook f32 -> f16 (row-major 512x128 == WMMA A-row layout already)
__global__ void k_cvt_h(const float* __restrict__ src, _Float16* __restrict__ dst, int n)
{
    int id = blockIdx.x * blockDim.x + threadIdx.x;
    if (id < n) dst[id] = (_Float16)src[id];
}

// ---- codebook row squared norms
__global__ void k_cnorm(const float* __restrict__ cb, float* __restrict__ cn, int K, int D)
{
    int k = blockIdx.x * blockDim.x + threadIdx.x;
    if (k >= K) return;
    float s = 0.0f;
    for (int d = 0; d < D; ++d) { float v = cb[(size_t)k * D + d]; s += v * v; }
    cn[k] = s;
}

// ---- argmin over codebook: d(n,k) = cnorm[k] - 2*G[n][k]  (||z||^2 constant in k)
__global__ void k_argmin(const float* __restrict__ G, const float* __restrict__ cn,
                         int* __restrict__ idx, int N, int K)
{
    int n = blockIdx.x * blockDim.x + threadIdx.x;
    if (n >= N) return;
    float best = 3.4e38f; int bi = 0;
    const float* row = G + (size_t)n * K;
    for (int k = 0; k < K; ++k) {
        float d = cn[k] - 2.0f * row[k];
        if (d < best) { best = d; bi = k; }
    }
    idx[n] = bi;
}

// ---- gather: e = codebook[idx]; f32 NCHW (d_out) + f16 NHWC (decoder input)
__global__ void k_gather_e(const float* __restrict__ cb, const int* __restrict__ idx,
                           float* __restrict__ e_nchw, _Float16* __restrict__ eh,
                           int N, int D, int HW)
{
    int id = blockIdx.x * blockDim.x + threadIdx.x;
    if (id >= N * D) return;
    int n = id / D, d = id - n * D;
    float v = cb[(size_t)idx[n] * D + d];
    eh[(size_t)n * D + d] = (_Float16)v;
    int pos = n % HW, b = n / HW;
    e_nchw[((size_t)b * D + d) * HW + pos] = v;
}

// ---------------------------------------------------------------------------
static inline size_t alignUp(size_t v, size_t a) { return (v + a - 1) & ~(a - 1); }

extern "C" void kernel_launch(void* const* d_in, const int* in_sizes, int n_in,
                              void* d_out, int out_size, void* d_ws, size_t ws_size,
                              hipStream_t stream)
{
    (void)in_sizes; (void)n_in; (void)out_size; (void)ws_size;

    const float* x        = (const float*)d_in[0];
    const float* enc_w1   = (const float*)d_in[1];
    const float* enc_b1   = (const float*)d_in[2];
    const float* bn1_g    = (const float*)d_in[3];
    const float* bn1_b    = (const float*)d_in[4];
    const float* bn1_m    = (const float*)d_in[5];
    const float* bn1_v    = (const float*)d_in[6];
    const float* enc_w2   = (const float*)d_in[7];
    const float* enc_b2   = (const float*)d_in[8];
    const float* bn2_g    = (const float*)d_in[9];
    const float* bn2_b    = (const float*)d_in[10];
    const float* bn2_m    = (const float*)d_in[11];
    const float* bn2_v    = (const float*)d_in[12];
    const float* enc_w3   = (const float*)d_in[13];
    const float* enc_b3   = (const float*)d_in[14];
    const float* codebook = (const float*)d_in[15];
    const float* dec_w1   = (const float*)d_in[16];
    const float* dec_b1   = (const float*)d_in[17];
    const float* dec_w2   = (const float*)d_in[18];
    const float* dec_b2   = (const float*)d_in[19];
    const float* dec_w3   = (const float*)d_in[20];
    const float* dec_b3   = (const float*)d_in[21];

    const int Bn = 8, L = 156, Lp = 160, H = 128, W = 128, D = 128, K = 512;
    const int H2 = 64, W2 = 64, H4 = 32, W4 = 32;
    const int Npos = Bn * H4 * W4; // 8192

    // ---- workspace carve
    char* ws = (char*)d_ws;
    size_t off = 0;
    auto carve = [&](size_t bytes) -> void* {
        void* p = ws + off; off = alignUp(off + bytes, 256); return p;
    };
    _Float16* xh   = (_Float16*)carve((size_t)Bn * H * W * Lp * 2);    // NHWC, C padded
    _Float16* w1h  = (_Float16*)carve((size_t)64  * (Lp * 16) * 2);
    _Float16* w2h  = (_Float16*)carve((size_t)128 * (64 * 16) * 2);
    _Float16* w3h  = (_Float16*)carve((size_t)128 * (128 * 9) * 2);
    _Float16* wd1h = (_Float16*)carve((size_t)128 * (128 * 9) * 2);
    _Float16* wt2h = (_Float16*)carve((size_t)64  * (128 * 16) * 2);
    _Float16* wt3h = (_Float16*)carve((size_t)160 * (64 * 16) * 2);    // 156 -> 160 rows
    _Float16* a1h  = (_Float16*)carve((size_t)Bn * H2 * W2 * 64 * 2);
    _Float16* a2h  = (_Float16*)carve((size_t)Bn * H4 * W4 * 128 * 2);
    _Float16* zh   = (_Float16*)carve((size_t)Npos * D * 2);
    _Float16* ch   = (_Float16*)carve((size_t)K * D * 2);
    float*    cn   = (float*)carve((size_t)K * 4);
    float*    G    = (float*)carve((size_t)Npos * K * 4);
    _Float16* eh   = (_Float16*)carve((size_t)Npos * D * 2);
    _Float16* h1h  = (_Float16*)carve((size_t)Npos * D * 2);
    _Float16* h2h  = (_Float16*)carve((size_t)Bn * H2 * W2 * 64 * 2);
    _Float16* zbuf = (_Float16*)carve((size_t)512 * 2);                // OOB zero page

    // ---- d_out layout: x_hat (f32 NCHW) | e (f32 NCHW) | idx (i32)
    float* xhat_out = (float*)d_out;
    float* e_out    = (float*)d_out + (size_t)Bn * L * H * W;
    int*   idx_out  = (int*)((float*)d_out + (size_t)Bn * L * H * W + (size_t)Bn * D * H4 * W4);

    auto blocks = [](size_t total, int bs) { return (unsigned)((total + bs - 1) / bs); };

    // ---- packs / converts
    k_zero_h<<<2, 256, 0, stream>>>(zbuf, 512);
    k_x_nhwc<<<blocks((size_t)Bn * H * W * Lp, 256), 256, 0, stream>>>(x, xh, Bn, L, Lp, H, W);
    k_pack_conv <<<blocks(64  * (size_t)(Lp * 16),  256), 256, 0, stream>>>(enc_w1, w1h, 64, 64, L, Lp, 4, 4);
    k_pack_conv <<<blocks(128 * (size_t)(64 * 16),  256), 256, 0, stream>>>(enc_w2, w2h, 128, 128, 64, 64, 4, 4);
    k_pack_conv <<<blocks(128 * (size_t)(128 * 9),  256), 256, 0, stream>>>(enc_w3, w3h, 128, 128, 128, 128, 3, 3);
    k_pack_conv <<<blocks(128 * (size_t)(128 * 9),  256), 256, 0, stream>>>(dec_w1, wd1h, 128, 128, 128, 128, 3, 3);
    k_pack_trans<<<blocks(64  * (size_t)(128 * 16), 256), 256, 0, stream>>>(dec_w2, wt2h, 64, 64, 128, 128, 4, 4);
    k_pack_trans<<<blocks(160 * (size_t)(64 * 16),  256), 256, 0, stream>>>(dec_w3, wt3h, 156, 160, 64, 64, 4, 4);
    k_cvt_h<<<blocks((size_t)K * D, 256), 256, 0, stream>>>(codebook, ch, K * D);
    k_cnorm<<<blocks(K, 256), 256, 0, stream>>>(codebook, cn, K, D);

    // ---- WMMA convs: grid = (Npix/64, ceil(Mpairs/2)), block = (32,2) = 2 waves
    auto grid = [&](int outH, int outW, int Mpad) {
        return dim3((unsigned)((Bn * outH * outW) / (16 * NT)),
                    (unsigned)((Mpad / 32 + 1) / 2));
    };
    const dim3 blk(32, 2);

    // encoder
    k_conv_wmma<4,4,2,1,0,0,160><<<grid(H2,W2,64), blk, 0, stream>>>(
        xh,  w1h, zbuf, enc_b1, bn1_g, bn1_b, bn1_m, bn1_v, a1h, nullptr,
        H,  W,  H2, W2, 64, 64);
    k_conv_wmma<4,4,2,1,0,0,64><<<grid(H4,W4,128), blk, 0, stream>>>(
        a1h, w2h, zbuf, enc_b2, bn2_g, bn2_b, bn2_m, bn2_v, a2h, nullptr,
        H2, W2, H4, W4, 128, 128);
    k_conv_wmma<3,3,1,1,0,2,128><<<grid(H4,W4,128), blk, 0, stream>>>(
        a2h, w3h, zbuf, enc_b3, nullptr, nullptr, nullptr, nullptr, zh, nullptr,
        H4, W4, H4, W4, 128, 128);

    // VQ: G = zf . codebook^T (1x1 "conv" == GEMM), argmin, gather
    k_conv_wmma<1,1,1,0,0,4,128><<<grid(H4,W4,512), blk, 0, stream>>>(
        zh, ch, zbuf, nullptr, nullptr, nullptr, nullptr, nullptr, nullptr, G,
        H4, W4, H4, W4, K, K);
    k_argmin<<<blocks(Npos, 256), 256, 0, stream>>>(G, cn, idx_out, Npos, K);
    k_gather_e<<<blocks((size_t)Npos * D, 256), 256, 0, stream>>>(
        codebook, idx_out, e_out, eh, Npos, D, H4 * W4);

    // decoder
    k_conv_wmma<3,3,1,1,0,1,128><<<grid(H4,W4,128), blk, 0, stream>>>(
        eh,  wd1h, zbuf, dec_b1, nullptr, nullptr, nullptr, nullptr, h1h, nullptr,
        H4, W4, H4, W4, 128, 128);
    k_conv_wmma<4,4,2,1,1,1,128><<<grid(H2,W2,64), blk, 0, stream>>>(
        h1h, wt2h, zbuf, dec_b2, nullptr, nullptr, nullptr, nullptr, h2h, nullptr,
        H4, W4, H2, W2, 64, 64);
    k_conv_wmma<4,4,2,1,1,3,64><<<grid(H,W,160), blk, 0, stream>>>(
        h2h, wt3h, zbuf, dec_b3, nullptr, nullptr, nullptr, nullptr, nullptr, xhat_out,
        H2, W2, H, W, 156, 160);
}
